// AttentionWithoutShortcutsLayer_84524956386049
// MI455X (gfx1250) — compile-verified, matrix operand order checked by reading, not run on hardware
//
#include <hip/hip_runtime.h>
#include <hip/hip_bf16.h>
#include <stdint.h>

#define GAMMA   1000.0f
#define HEADS   16
#define EDIM    64
#define DMODEL  1024
#define COEF    0.125f   // 1/sqrt(64)

typedef __attribute__((ext_vector_type(16))) __bf16        v16bf;
typedef __attribute__((ext_vector_type(8)))  float          v8f;
typedef __attribute__((ext_vector_type(8)))  unsigned short ushort8;

union FragBF {
    v16bf          v;
    ushort8        u2[2];
    unsigned short s[16];
};

__device__ __forceinline__ int lane_id() { return (int)(threadIdx.x & 31u); }

// native f32 -> bf16 (RNE) via hardware convert
__device__ __forceinline__ unsigned short f2bf(float f) {
    __bf16 h = (__bf16)f;
    return __builtin_bit_cast(unsigned short, h);
}

__device__ __forceinline__ v8f wmma_bf16(v16bf a, v16bf b, v8f c) {
    return __builtin_amdgcn_wmma_f32_16x16x32_bf16(
        /*neg_a=*/false, a, /*neg_b=*/false, b,
        /*c_mod=*/(short)0, c, /*reuse_a=*/false, /*reuse_b=*/false);
}

// 16x32 bf16 fragment whose per-lane "row" is contiguous along K in memory.
// Lane L: row = L&15, half = L>>4; chunk0 = k0+8*half .. +7, chunk1 = +16.
__device__ __forceinline__ v16bf frag_rows_bf16(const unsigned short* __restrict__ src,
                                                long ld, int k0) {
    int l = lane_id(), half = l >> 4;
    const unsigned short* p = src + (long)(l & 15) * ld + k0 + 8 * half;
    FragBF f;
    f.u2[0] = *(const ushort8*)p;
    f.u2[1] = *(const ushort8*)(p + 16);
    return f.v;
}

// ---------------- Kernel 0: bulk fp32 -> bf16 conversion ----------------
__global__ void __launch_bounds__(256)
cvt_bf16_kernel(const float* __restrict__ in,
                unsigned short* __restrict__ out, long n) {
    long i = ((long)blockIdx.x * blockDim.x + threadIdx.x) * 8;
    if (i >= n) return;
    float4 a = *(const float4*)(in + i);
    float4 b = *(const float4*)(in + i + 4);
    ushort8 u;
    u[0] = f2bf(a.x); u[1] = f2bf(a.y); u[2] = f2bf(a.z); u[3] = f2bf(a.w);
    u[4] = f2bf(b.x); u[5] = f2bf(b.y); u[6] = f2bf(b.z); u[7] = f2bf(b.w);
    *(ushort8*)(out + i) = u;
}

// ---------------- Kernel 1: per-head linear projection ----------------
// Wave computes 32(m) x 64(n). K-loop unrolled 2x with ping-pong fragment
// buffers: no rotation copies, each WMMA consumes fragments loaded a full
// pipeline stage (12 b128 loads) earlier.
__global__ void __launch_bounds__(256, 1)
qkv_proj_kernel(const unsigned short* __restrict__ X,  // [M,K] bf16
                const unsigned short* __restrict__ W,  // [N,K] bf16
                const float* __restrict__ bias,
                unsigned short* __restrict__ out,
                int M, int T, float scale, int transposedV) {
    const int N = DMODEL, K = DMODEL;
    int wave  = (int)((blockIdx.x * blockDim.x + threadIdx.x) >> 5);
    int nBlk  = wave & 15;          // N/64 = 16 column blocks
    int mTile = wave >> 4;          // M/32 tiles
    int m0 = mTile * 32, n0 = nBlk * 64;
    if (m0 >= M) return;
    int l = lane_id(), nl = l & 15, half = l >> 4;

    v8f acc[8];                     // [2 m-halves][4 n-tiles]
#pragma unroll
    for (int j = 0; j < 8; j++)
#pragma unroll
        for (int r = 0; r < 8; r++) acc[j][r] = 0.f;

    const unsigned short* X0 = X + (long)m0 * K;
    const unsigned short* X1 = X0 + (long)16 * K;

    v16bf a0A, a1A, bA[4], a0B, a1B, bB[4];
    auto loadf = [&](v16bf& x0, v16bf& x1, v16bf* bb, int kk) {
        x0 = frag_rows_bf16(X0, K, kk);
        x1 = frag_rows_bf16(X1, K, kk);
#pragma unroll
        for (int j = 0; j < 4; j++)
            bb[j] = frag_rows_bf16(W + (long)(n0 + 16 * j) * K, K, kk);
    };
    auto compute = [&](v16bf& x0, v16bf& x1, v16bf* bb) {
#pragma unroll
        for (int j = 0; j < 4; j++) {
            acc[j]     = wmma_bf16(x0, bb[j], acc[j]);
            acc[4 + j] = wmma_bf16(x1, bb[j], acc[4 + j]);
        }
    };

    const int steps = K / 32;       // 32 (even)
    loadf(a0A, a1A, bA, 0);
    int s = 0;
    for (; s + 2 < steps; s += 2) {
        __builtin_prefetch(X0 + (long)nl * K + (s + 2) * 32, 0, 3);
        loadf(a0B, a1B, bB, (s + 1) * 32);
        compute(a0A, a1A, bA);
        loadf(a0A, a1A, bA, (s + 2) * 32);
        compute(a0B, a1B, bB);
    }
    loadf(a0B, a1B, bB, (steps - 1) * 32);
    compute(a0A, a1A, bA);
    compute(a0B, a1B, bB);

    if (!transposedV) {
#pragma unroll
        for (int j = 0; j < 4; j++) {
            float bz = bias[n0 + 16 * j + nl];
#pragma unroll
            for (int r = 0; r < 8; r++) {
                out[(long)(m0 + r + 8 * half) * N + (n0 + 16 * j + nl)] =
                    f2bf((acc[j][r] + bz) * scale);
                out[(long)(m0 + 16 + r + 8 * half) * N + (n0 + 16 * j + nl)] =
                    f2bf((acc[4 + j][r] + bz) * scale);
            }
        }
    } else {
        int b = m0 / T, t0 = m0 - b * T;
#pragma unroll
        for (int j = 0; j < 4; j++) {
            int n = n0 + 16 * j + nl;
            int h = n >> 6, e = n & 63;
            float bz = bias[n];
            long rowoff = (long)((b * HEADS + h) * EDIM + e) * T;
            ushort8 u0, u1;
#pragma unroll
            for (int r = 0; r < 8; r++) {
                u0[r] = f2bf((acc[j][r] + bz) * scale);
                u1[r] = f2bf((acc[4 + j][r] + bz) * scale);
            }
            *(ushort8*)(out + rowoff + t0 + 8 * half)      = u0;
            *(ushort8*)(out + rowoff + t0 + 16 + 8 * half) = u1;
        }
    }
}

// ---------------- Kernel 2: causal flash attention (one 16-row q tile / wave) ----
__global__ void __launch_bounds__(256, 1)
attn_kernel(const unsigned short* __restrict__ Qb,  // [B,T,H*E] bf16 (COEF absorbed)
            const unsigned short* __restrict__ Kb,  // [B,T,H*E] bf16
            const unsigned short* __restrict__ Vt,  // [B,H,E,T] bf16
            unsigned short* __restrict__ PVt,       // [B,H,E,T] bf16
            int T) {
    __shared__ unsigned short Plds[8][16 * 32];
    int waveInBlk = (int)(threadIdx.x >> 5);
    int wave  = (int)(blockIdx.x * (blockDim.x >> 5)) + waveInBlk;
    int qTiles = T >> 4;
    int bh = wave / qTiles;
    int q0 = (wave - bh * qTiles) * 16;
    int b = bh >> 4, h = bh & 15;
    int l = lane_id(), nl = l & 15, half = l >> 4;

    const unsigned short* Qrow = Qb + (long)(b * T + q0) * DMODEL + h * EDIM;
    v16bf aQ0 = frag_rows_bf16(Qrow, DMODEL, 0);
    v16bf aQ1 = frag_rows_bf16(Qrow, DMODEL, 32);

    v8f acc[4];
    float mrun[8], lrun[8];
#pragma unroll
    for (int j = 0; j < 4; j++)
#pragma unroll
        for (int r = 0; r < 8; r++) acc[j][r] = 0.f;
#pragma unroll
    for (int r = 0; r < 8; r++) { mrun[r] = -3.0e38f; lrun[r] = 0.f; }

    const long vtBase = (long)bh * EDIM * T;

    for (int t0 = 0; t0 <= q0 + 15; t0 += 32) {
        // ---- issue ALL global fragment loads of this step up front ----
        v16bf kb[2][2], vb[4];
#pragma unroll
        for (int j = 0; j < 2; j++) {
            const unsigned short* Krow = Kb + (long)(b * T + t0 + 16 * j) * DMODEL + h * EDIM;
            kb[j][0] = frag_rows_bf16(Krow, DMODEL, 0);
            kb[j][1] = frag_rows_bf16(Krow, DMODEL, 32);
        }
#pragma unroll
        for (int j = 0; j < 4; j++)
            vb[j] = frag_rows_bf16(Vt + vtBase + (long)(16 * j) * T + t0, T, 0);

        v8f s[2];
#pragma unroll
        for (int j = 0; j < 2; j++) {
            v8f z;
#pragma unroll
            for (int r = 0; r < 8; r++) z[r] = 0.f;
            z = wmma_bf16(aQ0, kb[j][0], z);
            z = wmma_bf16(aQ1, kb[j][1], z);
            // causal mask: masked logits are exactly -GAMMA (reference semantics)
#pragma unroll
            for (int r = 0; r < 8; r++) {
                int qg = q0 + r + 8 * half;
                int tg = t0 + 16 * j + nl;
                z[r] = (tg > qg) ? -GAMMA : z[r];
            }
            s[j] = z;
        }
        // ---- online softmax (row lives across the 16 lanes of each half) ----
#pragma unroll
        for (int r = 0; r < 8; r++) {
            float tm = fmaxf(s[0][r], s[1][r]);
            for (int msk = 1; msk < 16; msk <<= 1) tm = fmaxf(tm, __shfl_xor(tm, msk));
            float mnew = fmaxf(mrun[r], tm);
            float corr = __expf(mrun[r] - mnew);
            float p0 = __expf(s[0][r] - mnew);
            float p1 = __expf(s[1][r] - mnew);
            s[0][r] = p0; s[1][r] = p1;
            float rs = p0 + p1;
            for (int msk = 1; msk < 16; msk <<= 1) rs += __shfl_xor(rs, msk);
            lrun[r] = lrun[r] * corr + rs;
            mrun[r] = mnew;
#pragma unroll
            for (int j = 0; j < 4; j++) acc[j][r] *= corr;
        }
        // ---- C-layout -> A-fragment layout transpose through LDS ----
#pragma unroll
        for (int j = 0; j < 2; j++)
#pragma unroll
            for (int r = 0; r < 8; r++)
                Plds[waveInBlk][(r + 8 * half) * 32 + 16 * j + nl] = f2bf(s[j][r]);

        FragBF fa;
#pragma unroll
        for (int i = 0; i < 8; i++) fa.s[i]     = Plds[waveInBlk][nl * 32 + 8 * half + i];
#pragma unroll
        for (int i = 0; i < 8; i++) fa.s[8 + i] = Plds[waveInBlk][nl * 32 + 16 + 8 * half + i];
        v16bf aP = fa.v;
#pragma unroll
        for (int j = 0; j < 4; j++) acc[j] = wmma_bf16(aP, vb[j], acc[j]);
    }

#pragma unroll
    for (int r = 0; r < 8; r++) {
        float inv = 1.0f / lrun[r];
#pragma unroll
        for (int j = 0; j < 4; j++) acc[j][r] *= inv;
    }
    // store PV transposed [B,H,E,T]: per lane, r runs over 8 consecutive t's
#pragma unroll
    for (int j = 0; j < 4; j++) {
        int e = 16 * j + nl;
        ushort8 u;
#pragma unroll
        for (int r = 0; r < 8; r++) u[r] = f2bf(acc[j][r]);
        *(ushort8*)(PVt + vtBase + (long)e * T + q0 + 8 * half) = u;
    }
}

// ---------------- Kernel 3: attn = D @ PV per (b,h) ----------------
// Wave computes 32(q) x 64(e); ping-pong pipelined over t.
__global__ void __launch_bounds__(256, 1)
mix_kernel(const unsigned short* __restrict__ Db,   // [T,T] bf16
           const unsigned short* __restrict__ PVt,  // [B,H,E,T] bf16
           unsigned short* __restrict__ Y,          // [B,T,H*E] bf16
           int T) {
    int wave  = (int)((blockIdx.x * blockDim.x + threadIdx.x) >> 5);
    int qTiles = T >> 5;            // 32-row q tiles
    int bh = wave / qTiles;
    int q0 = (wave - bh * qTiles) * 32;
    int b = bh >> 4, h = bh & 15;
    int l = lane_id(), nl = l & 15, half = l >> 4;

    v8f acc[8];
#pragma unroll
    for (int j = 0; j < 8; j++)
#pragma unroll
        for (int r = 0; r < 8; r++) acc[j][r] = 0.f;

    const unsigned short* D0 = Db + (long)q0 * T;
    const unsigned short* D1 = D0 + (long)16 * T;
    const long pvBase = (long)bh * EDIM * T;

    v16bf a0A, a1A, bA[4], a0B, a1B, bB[4];
    auto loadf = [&](v16bf& x0, v16bf& x1, v16bf* bb, int kk) {
        x0 = frag_rows_bf16(D0, T, kk);
        x1 = frag_rows_bf16(D1, T, kk);
#pragma unroll
        for (int j = 0; j < 4; j++)
            bb[j] = frag_rows_bf16(PVt + pvBase + (long)(16 * j) * T, T, kk);
    };
    auto compute = [&](v16bf& x0, v16bf& x1, v16bf* bb) {
#pragma unroll
        for (int j = 0; j < 4; j++) {
            acc[j]     = wmma_bf16(x0, bb[j], acc[j]);
            acc[4 + j] = wmma_bf16(x1, bb[j], acc[4 + j]);
        }
    };

    const int steps = T / 32;       // 64 (even)
    loadf(a0A, a1A, bA, 0);
    int s = 0;
    for (; s + 2 < steps; s += 2) {
        loadf(a0B, a1B, bB, (s + 1) * 32);
        compute(a0A, a1A, bA);
        loadf(a0A, a1A, bA, (s + 2) * 32);
        compute(a0B, a1B, bB);
    }
    loadf(a0B, a1B, bB, (steps - 1) * 32);
    compute(a0A, a1A, bA);
    compute(a0B, a1B, bB);

#pragma unroll
    for (int j = 0; j < 4; j++)
#pragma unroll
        for (int r = 0; r < 8; r++) {
            Y[(long)(b * T + q0 + r + 8 * half) * DMODEL + h * EDIM + 16 * j + nl] =
                f2bf(acc[j][r]);
            Y[(long)(b * T + q0 + 16 + r + 8 * half) * DMODEL + h * EDIM + 16 * j + nl] =
                f2bf(acc[4 + j][r]);
        }
}

// ---------------- Kernel 4: output projection (fp32 out), 32x64/wave, ping-pong ----
__global__ void __launch_bounds__(256, 1)
out_proj_kernel(const unsigned short* __restrict__ Yb, // [M,DMODEL] bf16
                const unsigned short* __restrict__ Wo, // [DMODEL,DMODEL] bf16
                const float* __restrict__ bo,
                float* __restrict__ out, int M) {
    const int N = DMODEL, K = DMODEL;
    int wave  = (int)((blockIdx.x * blockDim.x + threadIdx.x) >> 5);
    int nBlk  = wave & 15;
    int mTile = wave >> 4;
    int m0 = mTile * 32, n0 = nBlk * 64;
    if (m0 >= M) return;
    int l = lane_id(), nl = l & 15, half = l >> 4;

    v8f acc[8];
#pragma unroll
    for (int j = 0; j < 8; j++)
#pragma unroll
        for (int r = 0; r < 8; r++) acc[j][r] = 0.f;

    const unsigned short* Y0 = Yb + (long)m0 * K;
    const unsigned short* Y1 = Y0 + (long)16 * K;

    v16bf a0A, a1A, bA[4], a0B, a1B, bB[4];
    auto loadf = [&](v16bf& x0, v16bf& x1, v16bf* bb, int kk) {
        x0 = frag_rows_bf16(Y0, K, kk);
        x1 = frag_rows_bf16(Y1, K, kk);
#pragma unroll
        for (int j = 0; j < 4; j++)
            bb[j] = frag_rows_bf16(Wo + (long)(n0 + 16 * j) * K, K, kk);
    };
    auto compute = [&](v16bf& x0, v16bf& x1, v16bf* bb) {
#pragma unroll
        for (int j = 0; j < 4; j++) {
            acc[j]     = wmma_bf16(x0, bb[j], acc[j]);
            acc[4 + j] = wmma_bf16(x1, bb[j], acc[4 + j]);
        }
    };

    const int steps = K / 32;       // 32 (even)
    loadf(a0A, a1A, bA, 0);
    int s = 0;
    for (; s + 2 < steps; s += 2) {
        __builtin_prefetch(Y0 + (long)nl * K + (s + 2) * 32, 0, 3);
        loadf(a0B, a1B, bB, (s + 1) * 32);
        compute(a0A, a1A, bA);
        loadf(a0A, a1A, bA, (s + 2) * 32);
        compute(a0B, a1B, bB);
    }
    loadf(a0B, a1B, bB, (steps - 1) * 32);
    compute(a0A, a1A, bA);
    compute(a0B, a1B, bB);

#pragma unroll
    for (int j = 0; j < 4; j++) {
        float bz = bo[n0 + 16 * j + nl];
#pragma unroll
        for (int r = 0; r < 8; r++) {
            out[(long)(m0 + r + 8 * half) * N + n0 + 16 * j + nl]      = acc[j][r] + bz;
            out[(long)(m0 + 16 + r + 8 * half) * N + n0 + 16 * j + nl] = acc[4 + j][r] + bz;
        }
    }
}

extern "C" void kernel_launch(void* const* d_in, const int* in_sizes, int n_in,
                              void* d_out, int out_size, void* d_ws, size_t ws_size,
                              hipStream_t stream) {
    const float* q   = (const float*)d_in[0];
    const float* k   = (const float*)d_in[1];
    const float* v   = (const float*)d_in[2];
    const float* Wq  = (const float*)d_in[3];   // [H,E,D] == [H*E, D] row-major
    const float* bq  = (const float*)d_in[4];
    const float* Wk  = (const float*)d_in[5];
    const float* bk  = (const float*)d_in[6];
    const float* Wv  = (const float*)d_in[7];
    const float* bv  = (const float*)d_in[8];
    const float* Wo  = (const float*)d_in[9];
    const float* bo  = (const float*)d_in[10];
    const float* Dm  = (const float*)d_in[11];
    float* out = (float*)d_out;

    const int B = 4, T = 2048;
    const int M = B * T;                        // 8192 token rows
    const long elems = (long)M * DMODEL;        // 8.39M elems
    const long welems = (long)DMODEL * DMODEL;  // 1.05M elems
    const long delems = (long)T * T;            // 4.19M elems

    unsigned short* Xq  = (unsigned short*)d_ws;
    unsigned short* Xk  = Xq + elems;
    unsigned short* Xv  = Xk + elems;
    unsigned short* Qb  = Xv + elems;
    unsigned short* Kb  = Qb + elems;
    unsigned short* Vt  = Kb + elems;
    unsigned short* Wqb = Vt + elems;
    unsigned short* Wkb = Wqb + welems;
    unsigned short* Wvb = Wkb + welems;
    unsigned short* Wob = Wvb + welems;         // total ~104 MB
    // aliases (safe via stream-serial ordering):
    unsigned short* PVt = Xq;   // written after proj-Q consumed Xq
    unsigned short* Yb  = Xk;   // written after proj-K consumed Xk
    unsigned short* Db  = Xv;   // written after proj-V consumed Xv

    dim3 blk(256);
    // fp32 -> bf16 pre-conversion (removes all cvt work from GEMM loops)
    cvt_bf16_kernel<<<dim3((elems  / 8 + 255) / 256), blk, 0, stream>>>(q,  Xq,  elems);
    cvt_bf16_kernel<<<dim3((elems  / 8 + 255) / 256), blk, 0, stream>>>(k,  Xk,  elems);
    cvt_bf16_kernel<<<dim3((elems  / 8 + 255) / 256), blk, 0, stream>>>(v,  Xv,  elems);
    cvt_bf16_kernel<<<dim3((welems / 8 + 255) / 256), blk, 0, stream>>>(Wq, Wqb, welems);
    cvt_bf16_kernel<<<dim3((welems / 8 + 255) / 256), blk, 0, stream>>>(Wk, Wkb, welems);
    cvt_bf16_kernel<<<dim3((welems / 8 + 255) / 256), blk, 0, stream>>>(Wv, Wvb, welems);
    cvt_bf16_kernel<<<dim3((welems / 8 + 255) / 256), blk, 0, stream>>>(Wo, Wob, welems);

    int wavesProj = (M / 32) * (DMODEL / 64);  // 4096 waves (32x64 tiles)
    dim3 grdProj(wavesProj / 8);
    int wavesAttn = (B * HEADS) * (T / 16);    // 8192 waves
    dim3 grdAttn(wavesAttn / 8);
    int wavesMix  = (B * HEADS) * (T / 32);    // 4096 waves
    dim3 grdMix(wavesMix / 8);

    qkv_proj_kernel<<<grdProj, blk, 0, stream>>>(Xq, Wqb, bq, Qb, M, T, COEF, 0);
    qkv_proj_kernel<<<grdProj, blk, 0, stream>>>(Xk, Wkb, bk, Kb, M, T, 1.0f, 0);
    qkv_proj_kernel<<<grdProj, blk, 0, stream>>>(Xv, Wvb, bv, Vt, M, T, 1.0f, 1);
    // D -> bf16 into Xv's region (proj-V has consumed Xv by stream order)
    cvt_bf16_kernel<<<dim3((delems / 8 + 255) / 256), blk, 0, stream>>>(Dm, Db, delems);
    attn_kernel    <<<grdAttn, blk, 0, stream>>>(Qb, Kb, Vt, PVt, T);
    mix_kernel     <<<grdMix,  blk, 0, stream>>>(Db, PVt, Yb, T);
    out_proj_kernel<<<grdProj, blk, 0, stream>>>(Yb, Wob, bo, out, M);
}
